// OptimizedScaleShiftInvariantMACE_27109833572363
// MI455X (gfx1250) — matrix operand order repaired and checked by reading, not compiled, not run on hardware
//
#include <hip/hip_runtime.h>
#include <hip/hip_bf16.h>
#include <math.h>

#define NN 8192
#define NE 131072
#define NG 64

typedef __attribute__((ext_vector_type(16))) _Float16 v16h;
typedef __attribute__((ext_vector_type(8)))  _Float16 v8h;
typedef __attribute__((ext_vector_type(8)))  float    v8f;
typedef int v4i_t __attribute__((vector_size(16)));

#if __has_builtin(__builtin_amdgcn_global_load_async_to_lds_b128) && \
    __has_builtin(__builtin_amdgcn_s_wait_asynccnt)
#define USE_ASYNC_LDS 1
#endif

__device__ __forceinline__ int imin(int a, int b) { return a < b ? a : b; }

// 16-byte global -> LDS copy; on gfx1250 uses the async LDS-DMA path
// (GLOBAL_LOAD_ASYNC_TO_LDS_B128, tracked by ASYNCcnt).
__device__ __forceinline__ void stage_copy16(void* lds_dst, const void* gsrc) {
#ifdef USE_ASYNC_LDS
  __builtin_amdgcn_global_load_async_to_lds_b128((v4i_t*)gsrc, (v4i_t*)lds_dst, 0, 0);
#else
  *(uint4*)lds_dst = *(const uint4*)gsrc;
#endif
}

__device__ __forceinline__ void stage_fence_block() {
#ifdef USE_ASYNC_LDS
  __builtin_amdgcn_s_wait_asynccnt(0);
#endif
  __syncthreads();
}

// ---- WMMA fragment helpers (wave32, 16x16x32 f16 -> f32) ------------------
// A layout: lane holds row M=lane%16; lanes 0-15: K = k0+{0..7, 16..23},
// lanes 16-31: K = k0+{8..15, 24..31}.
__device__ __forceinline__ v16h load_a_f32(const float* __restrict__ rowp, int k0, int lane) {
  const float* p = rowp + k0 + ((lane >> 4) << 3);
  const float4* p4 = reinterpret_cast<const float4*>(p);
  float4 u0 = p4[0], u1 = p4[1], u2 = p4[4], u3 = p4[5];
  v16h a;
  a[0]=(_Float16)u0.x; a[1]=(_Float16)u0.y; a[2]=(_Float16)u0.z; a[3]=(_Float16)u0.w;
  a[4]=(_Float16)u1.x; a[5]=(_Float16)u1.y; a[6]=(_Float16)u1.z; a[7]=(_Float16)u1.w;
  a[8]=(_Float16)u2.x; a[9]=(_Float16)u2.y; a[10]=(_Float16)u2.z; a[11]=(_Float16)u2.w;
  a[12]=(_Float16)u3.x; a[13]=(_Float16)u3.y; a[14]=(_Float16)u3.z; a[15]=(_Float16)u3.w;
  return a;
}

__device__ __forceinline__ v16h load_a_f16lds(const _Float16* rowp, int k0, int lane) {
  const _Float16* p = rowp + k0 + ((lane >> 4) << 3);
  v8h lo = *(const v8h*)p;
  v8h hi = *(const v8h*)(p + 16);
  v16h a;
#pragma unroll
  for (int j = 0; j < 8; ++j) { a[j] = lo[j]; a[8 + j] = hi[j]; }
  return a;
}

// 16 gathered rows x 64 cols GEMM from pre-packed B fragments in LDS.
__device__ __forceinline__ void gemm_rows16(
    int rowA, const int* rowsC,
    const float* __restrict__ A, long lda, long aoff,
    const _Float16* sB,
    float* __restrict__ Out, long ldo, long ooff,
    const float* __restrict__ Add, long ldadd,
    float scale, int lane) {
  const float* arow = A + (long)rowA * lda + aoff;
  v16h a0 = load_a_f32(arow, 0, lane);
  v16h a1 = load_a_f32(arow, 32, lane);
#pragma unroll
  for (int nt = 0; nt < 4; ++nt) {
    v8f c = {};
    v16h b0 = *(const v16h*)(sB + ((nt * 2 + 0) * 32 + lane) * 16);
    v16h b1 = *(const v16h*)(sB + ((nt * 2 + 1) * 32 + lane) * 16);
    c = __builtin_amdgcn_wmma_f32_16x16x32_f16(false, a0, false, b0, (short)0, c, false, false);
    c = __builtin_amdgcn_wmma_f32_16x16x32_f16(false, a1, false, b1, (short)0, c, false, false);
    int col = nt * 16 + (lane & 15);
    if (Add) {
#pragma unroll
      for (int r = 0; r < 8; ++r) {
        long row = rowsC[r];
        Out[row * ldo + ooff + col] = c[r] * scale + Add[row * ldadd + col];
      }
    } else {
#pragma unroll
      for (int r = 0; r < 8; ++r) {
        long row = rowsC[r];
        Out[row * ldo + ooff + col] = c[r] * scale;
      }
    }
  }
}

// ---- weight pre-packing (once per call; packed f16 B-fragments in ws) -----
__global__ __launch_bounds__(256)
void k_pack64(const float* __restrict__ src, _Float16* __restrict__ dst, int src_stride) {
  int j = blockIdx.y;
  const float* W = src + (long)j * src_stride;
  _Float16* D = dst + (long)j * 4096;
  int idx = blockIdx.x * 256 + threadIdx.x;
  int hh = idx & 15, ln = (idx >> 4) & 31, ks = (idx >> 9) & 1, nt = idx >> 10;
  int k = ks * 32 + ((hh >> 3) << 4) + ((ln >> 4) << 3) + (hh & 7);
  int n = nt * 16 + (ln & 15);
  D[idx] = (_Float16)W[k * 64 + n];
}

__global__ __launch_bounds__(256)
void k_packr1(const float* __restrict__ Wr1, _Float16* __restrict__ dst) {
  int i = blockIdx.y;
  int idx = blockIdx.x * 256 + threadIdx.x;  // < 2048
  int hh = idx & 15, ln = (idx >> 4) & 31, nt = idx >> 9;
  int k = ((ln >> 4) << 4) + hh;
  int n = nt * 16 + (ln & 15);
  dst[i * 2048 + idx] = (_Float16)((k < 8) ? Wr1[i * 512 + k * 64 + n] : 0.f);
}

__global__ __launch_bounds__(256)
void k_packr2(const float* __restrict__ Wr2, _Float16* __restrict__ dst) {
  int i = blockIdx.y >> 2, lg = blockIdx.y & 3;
  int idx = blockIdx.x * 256 + threadIdx.x;  // < 4096
  int hh = idx & 15, ln = (idx >> 4) & 31, ks = (idx >> 9) & 1, nt = idx >> 10;
  int k = ks * 32 + ((hh >> 3) << 4) + ((ln >> 4) << 3) + (hh & 7);
  int n = lg * 64 + nt * 16 + (ln & 15);
  dst[(long)blockIdx.y * 4096 + idx] = (_Float16)Wr2[i * 16384 + k * 256 + n];
}

// ---- kernels ---------------------------------------------------------------
__global__ void k_zero(float* __restrict__ p, int n) {
  int i = blockIdx.x * blockDim.x + threadIdx.x;
  if (i < n) p[i] = 0.f;
}

__global__ void k_zero4(float4* __restrict__ p, int n4) {
  int i = blockIdx.x * blockDim.x + threadIdx.x;
  if (i < n4) p[i] = make_float4(0.f, 0.f, 0.f, 0.f);
}

__global__ __launch_bounds__(64)
void k_embed(const float* __restrict__ node_attrs, const int* __restrict__ batch,
             const float* __restrict__ W_embed, const float* __restrict__ atomic_E,
             int* __restrict__ elems, int* __restrict__ bucket, int* __restrict__ bcnt,
             float* __restrict__ nf, float* __restrict__ out) {
  int n = blockIdx.x, c = threadIdx.x;
  __shared__ int s_e;
  if (c == 0) {
    const float* a = node_attrs + (long)n * 10;
    int best = 0; float bv = a[0];
#pragma unroll
    for (int j = 1; j < 10; ++j) if (a[j] > bv) { bv = a[j]; best = j; }
    s_e = best;
    elems[n] = best;
    int pos = atomicAdd(&bcnt[best], 1);
    bucket[best * NN + pos] = n;
    atomicAdd(&out[batch[n]], atomic_E[best]);
  }
  __syncthreads();
  nf[(long)n * 64 + c] = W_embed[s_e * 64 + c];
}

__global__ __launch_bounds__(256)
void k_geom(const float* __restrict__ pos, const float* __restrict__ shifts,
            const int* __restrict__ ei, float* __restrict__ sh, _Float16* __restrict__ rbf16) {
  int e = blockIdx.x * blockDim.x + threadIdx.x;
  if (e >= NE) return;
  int s = ei[e], rcv = ei[NE + e];
  float x = pos[rcv * 3 + 0] - pos[s * 3 + 0] + shifts[e * 3 + 0];
  float y = pos[rcv * 3 + 1] - pos[s * 3 + 1] + shifts[e * 3 + 1];
  float z = pos[rcv * 3 + 2] - pos[s * 3 + 2] + shifts[e * 3 + 2];
  float r = sqrtf(x * x + y * y + z * z + 1e-18f);
  float ir = 1.f / r;
  x *= ir; y *= ir; z *= ir;
  const float s3 = 1.7320508075688772f, s5 = 2.2360679774997896f, s15 = 3.872983346207417f;
  const float ca = 2.0916500663351885f;  // sqrt(35/8)
  const float cb = 10.246950765959598f;  // sqrt(105)
  const float cc = 1.6201851746019651f;  // sqrt(21/8)
  const float cd = 1.3228756555322954f;  // sqrt(7/4)
  float* o = sh + (long)e * 16;
  float z2 = z * z;
  o[0] = 1.f;
  o[1] = s3 * x; o[2] = s3 * y; o[3] = s3 * z;
  o[4] = s15 * x * y; o[5] = s15 * y * z; o[6] = 0.5f * s5 * (3.f * z2 - 1.f);
  o[7] = s15 * x * z; o[8] = 0.5f * s15 * (x * x - y * y);
  o[9] = ca * y * (3.f * x * x - y * y);
  o[10] = cb * x * y * z;
  o[11] = cc * y * (5.f * z2 - 1.f);
  o[12] = cd * z * (5.f * z2 - 3.f);
  o[13] = cc * x * (5.f * z2 - 1.f);
  o[14] = 0.5f * cb * z * (x * x - y * y);
  o[15] = ca * x * (x * x - y * y);
  float rs = fmaxf(r, 1e-9f);
  float u = r / 5.0f;
  float u2 = u * u, u3 = u2 * u, u6 = u3 * u3;
  float fc = (u < 1.f) ? (1.f - 28.f * u6 + 48.f * u6 * u - 21.f * u6 * u2) : 0.f;
  float pref = sqrtf(2.f / 5.f) / rs * fc;
  float w = (float)M_PI * rs / 5.0f;
#pragma unroll
  for (int n = 1; n <= 8; ++n)
    rbf16[(long)e * 8 + n - 1] = (_Float16)(pref * __sinf((float)n * w));
}

__device__ __forceinline__ void stage_b_8k(_Float16* sB, const _Float16* PB, int tid, int nth) {
  for (int idx = tid; idx < 512; idx += nth) stage_copy16(sB + idx * 8, PB + idx * 8);
  stage_fence_block();
}

__global__ __launch_bounds__(128)
void k_dense64(const float* __restrict__ A, const _Float16* __restrict__ PB,
               float* __restrict__ Out, const float* __restrict__ Add) {
  __shared__ __align__(32) _Float16 sB[4096];
  int tid = threadIdx.x, lane = tid & 31, wave = tid >> 5;
  stage_b_8k(sB, PB, tid, 128);
  int tile = blockIdx.x * 4 + wave;
  int rowA = tile * 16 + (lane & 15);
  int rowsC[8];
#pragma unroll
  for (int r = 0; r < 8; ++r) rowsC[r] = tile * 16 + r + 8 * (lane >> 4);
  gemm_rows16(rowA, rowsC, A, 64, 0, sB, Out, 64, 0, Add, 64, 1.0f, lane);
}

__device__ __forceinline__ int l_of_lm(int lm) { return (lm == 0) ? 0 : (lm < 4) ? 1 : (lm < 9) ? 2 : 3; }

__global__ __launch_bounds__(128)
void k_lin(const float* __restrict__ m, const _Float16* __restrict__ PBase, float* __restrict__ m1) {
  __shared__ __align__(32) _Float16 sB[4096];
  int tid = threadIdx.x, lane = tid & 31, wave = tid >> 5;
  int lm = blockIdx.y;
  stage_b_8k(sB, PBase + (long)l_of_lm(lm) * 4096, tid, 128);
  int tile = blockIdx.x * 4 + wave;
  int rowA = tile * 16 + (lane & 15);
  int rowsC[8];
#pragma unroll
  for (int r = 0; r < 8; ++r) rowsC[r] = tile * 16 + r + 8 * (lane >> 4);
  gemm_rows16(rowA, rowsC, m, 1024, (long)lm * 64, sB, m1, 1024, (long)lm * 64,
              nullptr, 0, 1.0f / 16.0f, lane);
}

__global__ __launch_bounds__(128)
void k_skip0(const float* __restrict__ m1, const _Float16* __restrict__ PBase,
             float* __restrict__ m2, const int* __restrict__ bucket, const int* __restrict__ bcnt) {
  int e = blockIdx.z;
  int cnt = bcnt[e];
  if ((int)(blockIdx.x * 64) >= cnt) return;
  __shared__ __align__(32) _Float16 sB[4096];
  int tid = threadIdx.x, lane = tid & 31, wave = tid >> 5;
  int lm = blockIdx.y;
  stage_b_8k(sB, PBase + ((long)e * 4 + l_of_lm(lm)) * 4096, tid, 128);
  int tile = blockIdx.x * 4 + wave;
  const int* bk = bucket + (long)e * NN;
  int rowA = bk[imin(tile * 16 + (lane & 15), cnt - 1)];
  int rowsC[8];
#pragma unroll
  for (int r = 0; r < 8; ++r) rowsC[r] = bk[imin(tile * 16 + r + 8 * (lane >> 4), cnt - 1)];
  gemm_rows16(rowA, rowsC, m1, 1024, (long)lm * 64, sB, m2, 1024, (long)lm * 64,
              nullptr, 0, 1.0f, lane);
}

__global__ __launch_bounds__(128)
void k_skip1(const float* __restrict__ nf, const _Float16* __restrict__ PBase,
             float* __restrict__ sc, const int* __restrict__ bucket, const int* __restrict__ bcnt) {
  int e = blockIdx.z;
  int cnt = bcnt[e];
  if ((int)(blockIdx.x * 64) >= cnt) return;
  __shared__ __align__(32) _Float16 sB[4096];
  int tid = threadIdx.x, lane = tid & 31, wave = tid >> 5;
  stage_b_8k(sB, PBase + (long)e * 4096, tid, 128);
  int tile = blockIdx.x * 4 + wave;
  const int* bk = bucket + (long)e * NN;
  int rowA = bk[imin(tile * 16 + (lane & 15), cnt - 1)];
  int rowsC[8];
#pragma unroll
  for (int r = 0; r < 8; ++r) rowsC[r] = bk[imin(tile * 16 + r + 8 * (lane >> 4), cnt - 1)];
  gemm_rows16(rowA, rowsC, nf, 64, 0, sB, sc, 64, 0, nullptr, 0, 1.0f, lane);
}

// Fused radial MLP (rbf->silu->R, all WMMA) + message scatter; the 32MB
// accumulator stays L2-resident so the f32 atomics never touch HBM.
__global__ __launch_bounds__(256)
void k_msg(const _Float16* __restrict__ rbf16, const float* __restrict__ sh,
           const float* __restrict__ h, const int* __restrict__ ei,
           const _Float16* __restrict__ PWr1_i, const _Float16* __restrict__ PWr2_i,
           float* __restrict__ m) {
  __shared__ __align__(32) _Float16 sWr1[2048];   // 4 ntiles x 1 kstep
  __shared__ __align__(32) _Float16 sWr2[4096];   // 4 ntiles x 2 ksteps (per l-group)
  __shared__ __align__(32) _Float16 sH[8][1024];  // per-wave 16x64 f16
  __shared__ __align__(16) float    sR[8][1024];  // per-wave 16x64 f32
  __shared__ int sE[8][32];                       // per-wave sender[16], receiver[16]
  int tid = threadIdx.x, lane = tid & 31, wave = tid >> 5;
  int edge0 = blockIdx.x * 128 + wave * 16;
  // edge metadata into LDS
  if (lane < 16) sE[wave][lane] = ei[edge0 + lane];
  else           sE[wave][lane] = ei[NE + edge0 + (lane - 16)];
  // prefetch h[sender] rows (random gather) into cache ahead of the scatter
  {
    int snd_pf = ei[edge0 + (lane & 15)];
    __builtin_prefetch(h + (long)snd_pf * 64 + ((lane >> 4) << 5), 0, 3);
  }
  // stage Wr1 fragments (pre-packed)
  for (int idx = tid; idx < 256; idx += 256) stage_copy16(sWr1 + idx * 8, PWr1_i + idx * 8);
  stage_fence_block();
  // A fragment: 16 edges x 8 rbf features (K zero-padded to 32)
  v16h a = {};
  if (lane < 16) {
    const _Float16* p = rbf16 + (long)(edge0 + lane) * 8;
#pragma unroll
    for (int j = 0; j < 8; ++j) a[j] = p[j];
  }
  // H = silu(A @ Wr1) -> LDS (f16)
  _Float16* myH = sH[wave];
#pragma unroll
  for (int nt = 0; nt < 4; ++nt) {
    v8f c = {};
    v16h b = *(const v16h*)(sWr1 + (nt * 32 + lane) * 16);
    c = __builtin_amdgcn_wmma_f32_16x16x32_f16(false, a, false, b, (short)0, c, false, false);
    int col = nt * 16 + (lane & 15);
#pragma unroll
    for (int r = 0; r < 8; ++r) {
      float x = c[r];
      myH[(r + 8 * (lane >> 4)) * 64 + col] = (_Float16)(x / (1.f + __expf(-x)));
    }
  }
  v16h ha0 = load_a_f16lds(myH + (lane & 15) * 64, 0, lane);
  v16h ha1 = load_a_f16lds(myH + (lane & 15) * 64, 32, lane);
  const int LST[4] = {0, 1, 4, 9};
  const int LSZ[4] = {1, 3, 5, 7};
  float* myR = sR[wave];
  for (int lg = 0; lg < 4; ++lg) {
    __syncthreads();  // previous-iteration readers of sWr2 done
    for (int idx = tid; idx < 512; idx += 256)
      stage_copy16(sWr2 + idx * 8, PWr2_i + (long)lg * 4096 + idx * 8);
    stage_fence_block();
#pragma unroll
    for (int nt = 0; nt < 4; ++nt) {
      v8f c = {};
      v16h b0 = *(const v16h*)(sWr2 + ((nt * 2 + 0) * 32 + lane) * 16);
      v16h b1 = *(const v16h*)(sWr2 + ((nt * 2 + 1) * 32 + lane) * 16);
      c = __builtin_amdgcn_wmma_f32_16x16x32_f16(false, ha0, false, b0, (short)0, c, false, false);
      c = __builtin_amdgcn_wmma_f32_16x16x32_f16(false, ha1, false, b1, (short)0, c, false, false);
      int col = nt * 16 + (lane & 15);
#pragma unroll
      for (int r = 0; r < 8; ++r) myR[(r + 8 * (lane >> 4)) * 64 + col] = c[r];
    }
    int k0 = LST[lg], ksz = LSZ[lg];
    for (int e = 0; e < 16; ++e) {
      int edge = edge0 + e;
      int snd = sE[wave][e];
      int rcv = sE[wave][16 + e];
      float h0 = h[(long)snd * 64 + lane];
      float h1 = h[(long)snd * 64 + lane + 32];
      float r0 = myR[e * 64 + lane];
      float r1 = myR[e * 64 + lane + 32];
      float* mrow = m + (long)rcv * 1024;
      for (int kk = 0; kk < ksz; ++kk) {
        float s = sh[(long)edge * 16 + k0 + kk];
        atomicAdd(&mrow[(k0 + kk) * 64 + lane], s * r0 * h0);
        atomicAdd(&mrow[(k0 + kk) * 64 + lane + 32], s * r1 * h1);
      }
    }
  }
}

__global__ __launch_bounds__(64)
void k_inv(const float* __restrict__ m2, const int* __restrict__ elems,
           const float* __restrict__ Wc1_i, const float* __restrict__ Wc2_i,
           const float* __restrict__ Wc3_i, float* __restrict__ inv) {
  int n = blockIdx.x, c = threadIdx.x;
  const float* mr = m2 + (long)n * 1024;
  float x0 = mr[c];
  float p0 = x0 * x0, p1 = 0.f, p2 = 0.f, p3 = 0.f;
#pragma unroll
  for (int k = 1; k < 4; ++k) { float v = mr[k * 64 + c]; p1 += v * v; }
#pragma unroll
  for (int k = 4; k < 9; ++k) { float v = mr[k * 64 + c]; p2 += v * v; }
#pragma unroll
  for (int k = 9; k < 16; ++k) { float v = mr[k * 64 + c]; p3 += v * v; }
  int e = elems[n];
  const float* w2 = Wc2_i + (long)e * 256;
  const float* w3 = Wc3_i + (long)e * 256;
  float s2 = w2[c] * p0 + w2[64 + c] * p1 + w2[128 + c] * p2 + w2[192 + c] * p3;
  float s3 = w3[c] * p0 + w3[64 + c] * p1 + w3[128 + c] * p2 + w3[192 + c] * p3;
  float w1 = Wc1_i[(long)e * 256 + c];
  inv[(long)n * 64 + c] = w1 * x0 + s2 + s3 * x0;
}

__global__ __launch_bounds__(256)
void k_ro0(const float* __restrict__ nf, const float* __restrict__ W_ro0, float* __restrict__ site) {
  int n = blockIdx.x * blockDim.x + threadIdx.x;
  if (n >= NN) return;
  float acc = 0.f;
  const float* r = nf + (long)n * 64;
#pragma unroll
  for (int c = 0; c < 64; ++c) acc += r[c] * W_ro0[c];
  site[n] = acc;
}

__global__ __launch_bounds__(256)
void k_ro1(const float* __restrict__ nf, const float* __restrict__ Wa,
           const float* __restrict__ Wb, float* __restrict__ site) {
  int n = blockIdx.x * blockDim.x + threadIdx.x;
  if (n >= NN) return;
  float t[16];
#pragma unroll
  for (int j = 0; j < 16; ++j) t[j] = 0.f;
  const float* r = nf + (long)n * 64;
  for (int c = 0; c < 64; ++c) {
    float x = r[c];
#pragma unroll
    for (int j = 0; j < 16; ++j) t[j] += x * Wa[c * 16 + j];
  }
  float acc = 0.f;
#pragma unroll
  for (int j = 0; j < 16; ++j) acc += (t[j] / (1.f + __expf(-t[j]))) * Wb[j];
  site[n] += acc;
}

__global__ __launch_bounds__(256)
void k_final(const float* __restrict__ site, const int* __restrict__ batch,
             const float* __restrict__ scale, const float* __restrict__ shift,
             float* __restrict__ out) {
  int n = blockIdx.x * blockDim.x + threadIdx.x;
  if (n >= NN) return;
  atomicAdd(&out[batch[n]], scale[0] * site[n] + shift[0]);
}

// ---- host -----------------------------------------------------------------
extern "C" void kernel_launch(void* const* d_in, const int* in_sizes, int n_in,
                              void* d_out, int out_size, void* d_ws, size_t ws_size,
                              hipStream_t stream) {
  const float* positions = (const float*)d_in[0];
  const float* node_attrs = (const float*)d_in[1];
  const float* shifts = (const float*)d_in[2];
  const int* edge_index = (const int*)d_in[3];
  const int* batch = (const int*)d_in[4];
  const float* W_embed = (const float*)d_in[5];
  const float* atomic_E = (const float*)d_in[6];
  const float* Wr1 = (const float*)d_in[7];
  const float* Wr2 = (const float*)d_in[8];
  const float* W_up = (const float*)d_in[9];
  const float* W_lin = (const float*)d_in[10];
  const float* W_skip0 = (const float*)d_in[11];
  const float* W_skip1 = (const float*)d_in[12];
  const float* Wc1 = (const float*)d_in[13];
  const float* Wc2 = (const float*)d_in[14];
  const float* Wc3 = (const float*)d_in[15];
  const float* W_prod = (const float*)d_in[16];
  const float* W_ro0 = (const float*)d_in[17];
  const float* W_ro1a = (const float*)d_in[18];
  const float* W_ro1b = (const float*)d_in[19];
  const float* scale = (const float*)d_in[20];
  const float* shift = (const float*)d_in[21];
  float* out = (float*)d_out;

  char* ws = (char*)d_ws;
  size_t off = 0;
  auto alloc = [&](size_t bytes) -> void* {
    void* p = ws + off;
    off = (off + bytes + 255) & ~(size_t)255;
    return p;
  };
  float* m = (float*)alloc((size_t)NN * 1024 * 4);
  float* m1 = (float*)alloc((size_t)NN * 1024 * 4);
  float* m2 = (float*)alloc((size_t)NN * 1024 * 4);
  float* inv = (float*)alloc((size_t)NN * 64 * 4);
  float* sc = (float*)alloc((size_t)NN * 64 * 4);
  float* nf = (float*)alloc((size_t)NN * 64 * 4);
  float* h = (float*)alloc((size_t)NN * 64 * 4);
  float* shbuf = (float*)alloc((size_t)NE * 16 * 4);
  _Float16* rbf16 = (_Float16*)alloc((size_t)NE * 8 * 2);
  float* site = (float*)alloc((size_t)NN * 4);
  int* elems = (int*)alloc((size_t)NN * 4);
  int* bucket = (int*)alloc((size_t)10 * NN * 4);
  int* bcnt = (int*)alloc(64);
  // packed f16 B-fragment pool
  _Float16* PW = (_Float16*)alloc((size_t)290816 * 2);
  _Float16* PW_up = PW;               // 2 x 4096
  _Float16* PW_prod = PW + 8192;      // 2 x 4096
  _Float16* PW_lin = PW + 16384;      // 8 x 4096 (layer-major: [i][l])
  _Float16* PW_skip0 = PW + 49152;    // 40 x 4096 ([e][l])
  _Float16* PW_skip1 = PW + 212992;   // 10 x 4096
  _Float16* PWr1 = PW + 253952;       // 2 x 2048
  _Float16* PWr2 = PW + 258048;       // 8 x 4096 ([i][lg])

  // pre-pack all weight matrices into WMMA B-fragment layout (f16)
  k_pack64<<<dim3(16, 2), 256, 0, stream>>>(W_up, PW_up, 4096);
  k_pack64<<<dim3(16, 2), 256, 0, stream>>>(W_prod, PW_prod, 4096);
  k_pack64<<<dim3(16, 8), 256, 0, stream>>>(W_lin, PW_lin, 4096);
  k_pack64<<<dim3(16, 40), 256, 0, stream>>>(W_skip0, PW_skip0, 4096);
  k_pack64<<<dim3(16, 10), 256, 0, stream>>>(W_skip1, PW_skip1, 4096);
  k_packr1<<<dim3(8, 2), 256, 0, stream>>>(Wr1, PWr1);
  k_packr2<<<dim3(16, 8), 256, 0, stream>>>(Wr2, PWr2);

  k_zero<<<1, 64, 0, stream>>>(out, NG);
  k_zero<<<1, 32, 0, stream>>>((float*)bcnt, 10);
  k_embed<<<NN, 64, 0, stream>>>(node_attrs, batch, W_embed, atomic_E, elems, bucket, bcnt, nf, out);
  k_geom<<<NE / 256, 256, 0, stream>>>(positions, shifts, edge_index, shbuf, rbf16);

  for (int i = 0; i < 2; ++i) {
    if (i == 1)
      k_skip1<<<dim3(128, 1, 10), 128, 0, stream>>>(nf, PW_skip1, sc, bucket, bcnt);
    k_dense64<<<128, 128, 0, stream>>>(nf, PW_up + (long)i * 4096, h, nullptr);
    k_zero4<<<(NN * 1024 / 4) / 256, 256, 0, stream>>>((float4*)m, NN * 1024 / 4);
    k_msg<<<NE / 128, 256, 0, stream>>>(rbf16, shbuf, h, edge_index,
                                        PWr1 + (long)i * 2048, PWr2 + (long)i * 16384, m);
    k_lin<<<dim3(128, 16), 128, 0, stream>>>(m, PW_lin + (long)i * 16384, m1);
    const float* mm = m1;
    if (i == 0) {
      k_skip0<<<dim3(128, 16, 10), 128, 0, stream>>>(m1, PW_skip0, m2, bucket, bcnt);
      mm = m2;
    }
    k_inv<<<NN, 64, 0, stream>>>(mm, elems, Wc1 + (long)i * 2560, Wc2 + (long)i * 2560,
                                 Wc3 + (long)i * 2560, inv);
    k_dense64<<<128, 128, 0, stream>>>(inv, PW_prod + (long)i * 4096, nf, (i == 1) ? sc : nullptr);
    if (i == 0)
      k_ro0<<<NN / 256, 256, 0, stream>>>(nf, W_ro0, site);
    else
      k_ro1<<<NN / 256, 256, 0, stream>>>(nf, W_ro1a, W_ro1b, site);
  }
  k_final<<<NN / 256, 256, 0, stream>>>(site, batch, scale, shift, out);
}
